// MambaDecoder_71622874628163
// MI455X (gfx1250) — matrix operand order repaired and checked by reading, not compiled, not run on hardware
//
#include <hip/hip_runtime.h>
#include <hip/hip_bf16.h>
#include <math.h>

// ---------------------------------------------------------------------------
// Model constants (match reference)
// ---------------------------------------------------------------------------
static constexpr int Bb    = 32;
static constexpr int Tt    = 2048;
static constexpr int D_IN  = 512;
static constexpr int Dd    = 256;
static constexpr int OUTD  = 41;
static constexpr int Ll    = 3;
static constexpr int INNER = 512;      // 2*D
static constexpr int INNER2 = 1024;    // 2*INNER (xz width)

// ---------------------------------------------------------------------------
// Vector types for WMMA (wave32, gfx1250)
// ---------------------------------------------------------------------------
typedef __attribute__((ext_vector_type(16))) __bf16 bf16x16;
typedef __attribute__((ext_vector_type(8)))  __bf16 bf16x8;
typedef __attribute__((ext_vector_type(8)))  float  floatx8;
typedef __attribute__((ext_vector_type(4)))  unsigned short u16x4;

union FragBF {
    struct { bf16x8 lo, hi; } h;
    bf16x16 v;
};

__device__ __forceinline__ unsigned short f2bf(float f) {
    unsigned int u = __float_as_uint(f);
    unsigned int r = u + 0x7FFFu + ((u >> 16) & 1u);   // round-to-nearest-even
    return (unsigned short)(r >> 16);
}

__device__ __forceinline__ float silu_f(float x) {
    return x / (1.0f + __expf(-x));
}

// ---------------------------------------------------------------------------
// Generic bf16 WMMA GEMM:  C[M,N] = epilogue(A[M,K] @ W[K,N] + bias)
//  - Block = 256 threads = 8 waves.  Tile BM=128, BN=64, BK=32.
//  - Waves: 4 over M x 2 over N; each wave = 2x2 v_wmma_f32_16x16x32_bf16.
//  - Software-pipelined staging: the 6 float4 global loads for tile k+1 are
//    issued right after the barrier, the 4 WMMAs for tile k execute while the
//    loads are in flight, and the wait/convert/LDS-store happens at the top
//    of the next iteration.  (Weights are L2-resident on MI455X; the A tile
//    streams from HBM, so per-wave latency hiding is what matters.)
//  - A tile staged in LDS as bf16 rows; W tile staged TRANSPOSED (N-major) so
//    both fragment loads are contiguous, 16B-aligned LDS reads matching the
//    wave32 WMMA VGPR layouts (lane halves carry K {0-7,16-23} / {8-15,24-31}).
//  - day_ids != nullptr => per-batch weight/bias (adapter), blockIdx.z = batch.
//  - Requires M%128==0, N%64==0, K%32==0 (true for every call site here).
// ---------------------------------------------------------------------------
static constexpr int BM = 128, BN = 64, BK = 32;
static constexpr int LDS_S = 40;   // padded LDS row stride (ushorts) -> conflict-free

__global__ __launch_bounds__(256)
void gemm_bf16_wmma(const float* __restrict__ A, const float* __restrict__ W,
                    const float* __restrict__ bias, float* __restrict__ C,
                    int M, int N, int K, int lda, int ldw, int ldc,
                    int epilogue /*0=none,1=softsign*/,
                    const int* __restrict__ day_ids, int bias_per_day)
{
    __shared__ unsigned short As[BM * LDS_S];
    __shared__ unsigned short Bs[BN * LDS_S];

    if (day_ids) {
        int d = day_ids[blockIdx.z];
        W += (size_t)d * K * ldw;
        if (bias_per_day && bias) bias += (size_t)d * N;
        A += (size_t)blockIdx.z * (size_t)M * lda;
        C += (size_t)blockIdx.z * (size_t)M * ldc;
    }

    const int tile_m = blockIdx.y * BM;
    const int tile_n = blockIdx.x * BN;
    const int tid  = threadIdx.x;
    const int lane = tid & 31;
    const int wid  = tid >> 5;
    const int half = lane >> 4;     // K-half selector per WMMA layout
    const int lr   = lane & 15;
    const int wm   = wid & 3;       // wave row (0..3) -> 32 rows of M
    const int wn   = wid >> 2;      // wave col (0..1) -> 32 cols of N

    // Loop-invariant staging coordinates.
    const int ar0 = tid >> 3;       // A rows: ar0 + 32*j   (j=0..3)
    const int ac4 = (tid & 7) << 2; // A col within BK
    const int wk0 = tid >> 4;       // W k rows: wk0 + 16*j (j=0..1)
    const int wc4 = (tid & 15) << 2;// W n col within BN

    floatx8 acc[2][2] = {};
    float4 ra[4];                   // staged A chunks (next tile)
    float4 rw[2];                   // staged W chunks (next tile)

    auto issue_loads = [&](int k0) {
        #pragma unroll
        for (int j = 0; j < 4; ++j)
            ra[j] = *(const float4*)(A + (size_t)(tile_m + ar0 + 32 * j) * lda + k0 + ac4);
        #pragma unroll
        for (int j = 0; j < 2; ++j)
            rw[j] = *(const float4*)(W + (size_t)(k0 + wk0 + 16 * j) * ldw + tile_n + wc4);
    };
    auto convert_store = [&]() {
        #pragma unroll
        for (int j = 0; j < 4; ++j) {
            u16x4 p = { f2bf(ra[j].x), f2bf(ra[j].y), f2bf(ra[j].z), f2bf(ra[j].w) };
            *(u16x4*)(&As[(ar0 + 32 * j) * LDS_S + ac4]) = p;
        }
        #pragma unroll
        for (int j = 0; j < 2; ++j) {
            int kr = wk0 + 16 * j;
            Bs[(wc4 + 0) * LDS_S + kr] = f2bf(rw[j].x);
            Bs[(wc4 + 1) * LDS_S + kr] = f2bf(rw[j].y);
            Bs[(wc4 + 2) * LDS_S + kr] = f2bf(rw[j].z);
            Bs[(wc4 + 3) * LDS_S + kr] = f2bf(rw[j].w);
        }
    };

    issue_loads(0);                 // prologue: stage tile 0

    for (int k0 = 0; k0 < K; k0 += BK) {
        convert_store();            // waits outstanding loads, fills LDS
        __syncthreads();

        if (k0 + BK < K)
            issue_loads(k0 + BK);   // in flight during the WMMAs below

        FragBF fa[2], fb[2];
        #pragma unroll
        for (int i = 0; i < 2; ++i) {
            int row = (wm * 32 + i * 16 + lr) * LDS_S;
            int c0  = half * 8;     // lanes 0-15: K{0-7,16-23}; 16-31: K{8-15,24-31}
            fa[i].h.lo = *(const bf16x8*)(&As[row + c0]);
            fa[i].h.hi = *(const bf16x8*)(&As[row + c0 + 16]);
        }
        #pragma unroll
        for (int j = 0; j < 2; ++j) {
            int row = (wn * 32 + j * 16 + lr) * LDS_S;
            int kc  = half * 8;     // lanes 0-15: K{0-7,16-23}; 16-31: K{8-15,24-31}
            fb[j].h.lo = *(const bf16x8*)(&Bs[row + kc]);
            fb[j].h.hi = *(const bf16x8*)(&Bs[row + kc + 16]);
        }
        #pragma unroll
        for (int i = 0; i < 2; ++i)
            #pragma unroll
            for (int j = 0; j < 2; ++j)
                acc[i][j] = __builtin_amdgcn_wmma_f32_16x16x32_bf16(
                    false, fa[i].v, false, fb[j].v, (short)0, acc[i][j], false, false);
        __syncthreads();
    }

    // ---- epilogue + store (C/D layout: VGPR r -> M = half*8 + r, lane -> N) ----
    #pragma unroll
    for (int i = 0; i < 2; ++i) {
        #pragma unroll
        for (int j = 0; j < 2; ++j) {
            int n = tile_n + wn * 32 + j * 16 + lr;
            float bv = bias ? bias[n] : 0.0f;
            #pragma unroll
            for (int r = 0; r < 8; ++r) {
                int m = tile_m + wm * 32 + i * 16 + half * 8 + r;
                float v = acc[i][j][r] + bv;
                if (epilogue == 1) v = v / (1.0f + fabsf(v));   // soft_sign
                C[(size_t)m * ldc + n] = v;
            }
        }
    }
}

// ---------------------------------------------------------------------------
// LayerNorm over D=256, one wave (32 lanes) per row. Optional residual + SiLU.
// ---------------------------------------------------------------------------
__global__ __launch_bounds__(256)
void ln_act(const float* __restrict__ x, const float* __restrict__ res,
            const float* __restrict__ s, const float* __restrict__ b,
            float* __restrict__ out, int nrows, int do_silu)
{
    int gt   = blockIdx.x * blockDim.x + threadIdx.x;
    int row  = gt >> 5;
    int lane = gt & 31;
    if (row >= nrows) return;
    const size_t base = (size_t)row * Dd;

    float v[8];
    float sum = 0.0f;
    #pragma unroll
    for (int e = 0; e < 8; ++e) {
        int c = lane + 32 * e;
        float t = x[base + c];
        if (res) t += res[base + c];
        v[e] = t; sum += t;
    }
    #pragma unroll
    for (int o = 16; o > 0; o >>= 1) sum += __shfl_xor(sum, o, 32);
    float mean = sum * (1.0f / Dd);

    float var = 0.0f;
    #pragma unroll
    for (int e = 0; e < 8; ++e) { float d = v[e] - mean; var += d * d; }
    #pragma unroll
    for (int o = 16; o > 0; o >>= 1) var += __shfl_xor(var, o, 32);
    float rstd = rsqrtf(var * (1.0f / Dd) + 1e-5f);

    #pragma unroll
    for (int e = 0; e < 8; ++e) {
        int c = lane + 32 * e;
        float y = (v[e] - mean) * rstd * s[c] + b[c];
        if (do_silu) y = silu_f(y);
        out[base + c] = y;
    }
}

// ---------------------------------------------------------------------------
// Depthwise conv (k=4) + SiLU.  xi = xz[..., 0:512] (row stride 1024).
// dir=0 (forward, padding (2,1)):  y[t] = sum_k xi[t-2+k] * K[k]
// dir=1 (time-reversed branch, kept in physical order): y[t] = sum_k xi[t+2-k]*K[k]
// ---------------------------------------------------------------------------
__global__ __launch_bounds__(256)
void dwconv_silu_k(const float* __restrict__ xz, const float* __restrict__ Kw,
                   const float* __restrict__ Kb, float* __restrict__ xc, int dir)
{
    size_t idx = (size_t)blockIdx.x * blockDim.x + threadIdx.x;
    if (idx >= (size_t)Bb * Tt * INNER) return;
    int c = (int)(idx % INNER);
    size_t bt = idx / INNER;
    int t = (int)(bt % Tt);
    size_t brow = (bt / Tt) * (size_t)Tt;

    float k0 = Kw[c * 4 + 0], k1 = Kw[c * 4 + 1], k2 = Kw[c * 4 + 2], k3 = Kw[c * 4 + 3];
    float y = Kb[c];
    auto xi = [&](int tt) -> float {
        return (tt < 0 || tt >= Tt) ? 0.0f : xz[(brow + tt) * INNER2 + c];
    };
    if (dir == 0)
        y += k0 * xi(t - 2) + k1 * xi(t - 1) + k2 * xi(t) + k3 * xi(t + 1);
    else
        y += k0 * xi(t + 2) + k1 * xi(t + 1) + k2 * xi(t) + k3 * xi(t - 1);
    xc[idx] = silu_f(y);
}

// ---------------------------------------------------------------------------
// Per-(batch,channel) sequential scan over T:
//   dt = softplus(dtraw); c = cumsum(dt) (prefix fwd / suffix bwd);
//   gated = xc * exp(-0.1*c) * silu(z).   In-place over dtraw.
// 16384 independent scans; fully coalesced (consecutive c -> consecutive addr).
// ---------------------------------------------------------------------------
__global__ __launch_bounds__(256)
void scan_gate_k(float* __restrict__ dt_g, const float* __restrict__ xc,
                 const float* __restrict__ xz, int dir)
{
    int idx = blockIdx.x * blockDim.x + threadIdx.x;
    if (idx >= Bb * INNER) return;
    int c = idx % INNER;
    int b = idx / INNER;
    float accv = 0.0f;
    for (int s = 0; s < Tt; ++s) {
        int t = (dir == 0) ? s : (Tt - 1 - s);
        size_t off = (size_t)b * Tt + t;
        float d  = dt_g[off * INNER + c];
        float sp = (d > 20.0f) ? d : log1pf(__expf(d));   // softplus
        accv += sp;
        float w = __expf(-0.1f * accv);
        float z = xz[off * INNER2 + INNER + c];
        dt_g[off * INNER + c] = xc[off * INNER + c] * w * silu_f(z);
    }
}

// ---------------------------------------------------------------------------
// Output head: logits = h @ outW(256x41) + outb ; log_softmax over 41.
// One 64-thread block per row (N=41 too skinny for WMMA tiles).
// ---------------------------------------------------------------------------
__global__ __launch_bounds__(64)
void head_lsm(const float* __restrict__ h, const float* __restrict__ Wo,
              const float* __restrict__ bo, float* __restrict__ out)
{
    __shared__ float hrow[Dd];
    __shared__ float logits[48];
    __shared__ float red[2];
    size_t row = blockIdx.x;
    int tid = threadIdx.x;
    #pragma unroll
    for (int e = 0; e < 4; ++e) hrow[tid + 64 * e] = h[row * Dd + tid + 64 * e];
    __syncthreads();
    if (tid < OUTD) {
        float acc = bo[tid];
        for (int d = 0; d < Dd; ++d) acc += hrow[d] * Wo[d * OUTD + tid];
        logits[tid] = acc;
    }
    __syncthreads();
    if (tid == 0) {
        float mx = -1e30f;
        for (int o = 0; o < OUTD; ++o) mx = fmaxf(mx, logits[o]);
        float sm = 0.0f;
        for (int o = 0; o < OUTD; ++o) sm += __expf(logits[o] - mx);
        red[0] = mx; red[1] = logf(sm);
    }
    __syncthreads();
    if (tid < OUTD) out[row * OUTD + tid] = logits[tid] - red[0] - red[1];
}

// ---------------------------------------------------------------------------
// Host-side orchestration
// ---------------------------------------------------------------------------
extern "C" void kernel_launch(void* const* d_in, const int* in_sizes, int n_in,
                              void* d_out, int out_size, void* d_ws, size_t ws_size,
                              hipStream_t stream)
{
    (void)in_sizes; (void)n_in; (void)out_size; (void)ws_size;

    const float* x         = (const float*)d_in[0];
    const int*   day_ids   = (const int*)  d_in[1];
    const float* adapter_W = (const float*)d_in[2];
    const float* adapter_b = (const float*)d_in[3];
    const float* cW1       = (const float*)d_in[4];
    const float* cb1       = (const float*)d_in[5];
    const float* cln1_s    = (const float*)d_in[6];
    const float* cln1_b    = (const float*)d_in[7];
    const float* cW2       = (const float*)d_in[8];
    const float* cb2       = (const float*)d_in[9];
    const float* cln2_s    = (const float*)d_in[10];
    const float* cln2_b    = (const float*)d_in[11];
    const float* m_in      = (const float*)d_in[12];
    const float* m_convK   = (const float*)d_in[13];
    const float* m_convB   = (const float*)d_in[14];
    const float* m_dtW     = (const float*)d_in[15];
    const float* m_dtB     = (const float*)d_in[16];
    const float* m_out     = (const float*)d_in[17];
    const float* proj_W    = (const float*)d_in[18];
    const float* proj_b    = (const float*)d_in[19];
    const float* norm_s    = (const float*)d_in[20];
    const float* norm_b    = (const float*)d_in[21];
    const float* outW      = (const float*)d_in[22];
    const float* outb      = (const float*)d_in[23];
    float*       out       = (float*)d_out;

    const size_t BT = (size_t)Bb * Tt;

    // workspace layout (floats)
    float* bufA = (float*)d_ws;                 // BT*512  adapter out, later "combined"
    float* h256 = bufA + BT * 512;              // BT*256  current hidden state
    float* tmp  = h256 + BT * 256;              // BT*256  pre-LN gemm output
    float* xz   = tmp  + BT * 256;              // BT*1024 Win output
    float* xc   = xz   + BT * 1024;             // BT*512  conv+silu output
    float* dtb  = xc   + BT * 512;              // BT*512  dt raw -> gated (in place)

    const dim3 blk(256);
    const int  lnGrid = (int)(BT / 8);                     // 1 wave per row
    const int  ewGrid = (int)((BT * INNER + 255) / 256);   // elementwise over B*T*512

    // 1) per-day adapter: softsign(x @ W[day] + b[day]) -> bufA
    gemm_bf16_wmma<<<dim3(D_IN / BN, Tt / BM, Bb), blk, 0, stream>>>(
        x, adapter_W, adapter_b, bufA, Tt, D_IN, D_IN, D_IN, D_IN, D_IN,
        /*softsign*/1, day_ids, /*bias_per_day*/1);

    // 2) h = silu(LN(bufA @ cW1 + cb1))
    gemm_bf16_wmma<<<dim3(Dd / BN, BT / BM), blk, 0, stream>>>(
        bufA, cW1, cb1, tmp, (int)BT, Dd, D_IN, D_IN, Dd, Dd, 0, nullptr, 0);
    ln_act<<<lnGrid, blk, 0, stream>>>(tmp, nullptr, cln1_s, cln1_b, h256, (int)BT, 1);

    // 3) h = silu(LN(h @ cW2 + cb2))
    gemm_bf16_wmma<<<dim3(Dd / BN, BT / BM), blk, 0, stream>>>(
        h256, cW2, cb2, tmp, (int)BT, Dd, Dd, Dd, Dd, Dd, 0, nullptr, 0);
    ln_act<<<lnGrid, blk, 0, stream>>>(tmp, nullptr, cln2_s, cln2_b, h256, (int)BT, 1);

    // 4) bidirectional soft-window layers
    for (int l = 0; l < Ll; ++l) {
        for (int dir = 0; dir < 2; ++dir) {
            const size_t ld = (size_t)(l * 2 + dir);
            const float* Win  = m_in    + ld * (size_t)Dd * INNER2;
            const float* cK   = m_convK + ld * (size_t)INNER * 4;
            const float* cB   = m_convB + ld * (size_t)INNER;
            const float* dtW  = m_dtW   + ld * (size_t)INNER * INNER;
            const float* dtBp = m_dtB   + ld * (size_t)INNER;
            const float* Wout = m_out   + ld * (size_t)INNER * Dd;

            // xz = h @ Win
            gemm_bf16_wmma<<<dim3(INNER2 / BN, BT / BM), blk, 0, stream>>>(
                h256, Win, nullptr, xz, (int)BT, INNER2, Dd, Dd, INNER2, INNER2,
                0, nullptr, 0);
            // xc = silu(dwconv(xi))
            dwconv_silu_k<<<ewGrid, blk, 0, stream>>>(xz, cK, cB, xc, dir);
            // dtraw = xc @ dtW + dtB
            gemm_bf16_wmma<<<dim3(INNER / BN, BT / BM), blk, 0, stream>>>(
                xc, dtW, dtBp, dtb, (int)BT, INNER, INNER, INNER, INNER, INNER,
                0, nullptr, 0);
            // gated = xc * exp(-0.1*cumsum(softplus(dtraw))) * silu(z)   (in place)
            scan_gate_k<<<(Bb * INNER + 255) / 256, blk, 0, stream>>>(dtb, xc, xz, dir);
            // combined[:, dir*256 : dir*256+256] = gated @ Wout   (ldc=512)
            gemm_bf16_wmma<<<dim3(Dd / BN, BT / BM), blk, 0, stream>>>(
                dtb, Wout, nullptr, bufA + dir * Dd, (int)BT, Dd, INNER,
                INNER, Dd, 2 * Dd, 0, nullptr, 0);
        }
        // h = LN(h + combined @ proj_W + proj_b)
        gemm_bf16_wmma<<<dim3(Dd / BN, BT / BM), blk, 0, stream>>>(
            bufA, proj_W + (size_t)l * 2 * Dd * Dd, proj_b + (size_t)l * Dd, tmp,
            (int)BT, Dd, 2 * Dd, 2 * Dd, Dd, Dd, 0, nullptr, 0);
        ln_act<<<lnGrid, blk, 0, stream>>>(
            tmp, h256, norm_s + (size_t)l * Dd, norm_b + (size_t)l * Dd,
            h256, (int)BT, 0);
    }

    // 5) log_softmax(h @ outW + outb)
    head_lsm<<<(unsigned)BT, dim3(64), 0, stream>>>(h256, outW, outb, out);
}
